// MixedFeedForward_50826642980981
// MI455X (gfx1250) — matrix operand order repaired and checked by reading, not compiled, not run on hardware
//
#include <hip/hip_runtime.h>
#include <hip/hip_bf16.h>
#include <math.h>

#define D_MODEL 1024
#define D_FF    4096
#define SEQ_TOK 1024
#define NS_TOK  16
#define BATCH   8
#define SEQ_LEN (SEQ_TOK + NS_TOK)   // 1040
#define M_SEQ   (BATCH * SEQ_TOK)    // 8192

typedef __bf16 bf16_t;
typedef __attribute__((ext_vector_type(16))) __bf16 v16bf;
typedef __attribute__((ext_vector_type(8)))  float  v8f;

#define LDA 40   // padded halfword stride (80B rows, 16B aligned)

// CDNA5 async global->LDS path (ASYNCcnt) if the toolchain exposes it.
#if defined(__has_builtin)
#if __has_builtin(__builtin_amdgcn_global_load_async_to_lds_b128) && \
    __has_builtin(__builtin_amdgcn_s_wait_asynccnt)
#define HAVE_ASYNC_LDS 1
#endif
#endif

#ifdef HAVE_ASYNC_LDS
// Builtin signature (from clang diagnostic): arg0 = int4 AS(1)* (global src),
// arg1 = int4 AS(3)* (LDS dst), arg2/arg3 = imm offset / cpol.
typedef int v4i_t __attribute__((__vector_size__(4 * sizeof(int))));
typedef __attribute__((address_space(1))) v4i_t* gv4_t;
typedef __attribute__((address_space(3))) v4i_t* lv4_t;
#endif

__device__ __forceinline__ float gelu_exact(float v) {
    return 0.5f * v * (1.0f + erff(v * 0.70710678118654752440f));
}

// ---------------------------------------------------------------------------
// GEMM1: H[m,f] = GELU( x_seq[m,:] @ W1[:,f] + b1[f] ), H stored bf16.
// Block tile 128x128, 8 waves (4M x 2N), wave tile 32x64 -> 8 WMMA accs/wave.
// Software-pipelined: tile k+1 prefetched to registers during WMMAs of tile k.
// ---------------------------------------------------------------------------
__global__ __launch_bounds__(256)
void ffn_gemm1(const float* __restrict__ x, const float* __restrict__ W1,
               const float* __restrict__ b1, bf16_t* __restrict__ H)
{
    __shared__ bf16_t lsA[128 * LDA];   // [m][k] bf16
    __shared__ bf16_t lsB[128 * LDA];   // [n][k] bf16 (transposed)

    const int t    = threadIdx.x;
    const int lane = t & 31;
    const int wave = t >> 5;
    const int wm   = wave & 3;
    const int wn   = wave >> 2;
    const int half = lane >> 4;
    const int l16  = lane & 15;
    const int bm   = blockIdx.y * 128;
    const int bn   = blockIdx.x * 128;

    // staging ownership: A -> one row, 16-wide k half; B -> one n column, 16-wide k half
    const int arow = t >> 1;
    const int acb  = (t & 1) * 16;
    const int bcol = t & 127;
    const int bkh  = (t >> 7) * 16;

    const int xr = ((bm + arow) >> 10) * SEQ_LEN + ((bm + arow) & 1023);
    const float* aseg = x  + (size_t)xr * D_MODEL + acb;            // + k0
    const float* bseg = W1 + (size_t)bkh * D_FF + bn + bcol;        // + k0*D_FF

    v8f acc[2][4];
    #pragma unroll
    for (int i = 0; i < 2; ++i)
        #pragma unroll
        for (int j = 0; j < 4; ++j) acc[i][j] = (v8f){0,0,0,0,0,0,0,0};

    float areg[16], breg[16];
    // prologue: tile 0
    #pragma unroll
    for (int i = 0; i < 4; ++i) {
        float4 v = *(const float4*)(aseg + i * 4);
        areg[4*i+0] = v.x; areg[4*i+1] = v.y; areg[4*i+2] = v.z; areg[4*i+3] = v.w;
    }
    #pragma unroll
    for (int j = 0; j < 16; ++j) breg[j] = bseg[(size_t)j * D_FF];

    for (int k0 = 0; k0 < D_MODEL; k0 += 32) {
        // ---- convert + packed LDS stores (2 x ds_store_b128 each) ----
        union Pack { bf16_t h[16]; uint4 q[2]; } pa, pb;
        #pragma unroll
        for (int i = 0; i < 16; ++i) pa.h[i] = (bf16_t)areg[i];
        #pragma unroll
        for (int i = 0; i < 16; ++i) pb.h[i] = (bf16_t)breg[i];
        { uint4* d = (uint4*)&lsA[arow * LDA + acb]; d[0] = pa.q[0]; d[1] = pa.q[1]; }
        { uint4* d = (uint4*)&lsB[bcol * LDA + bkh]; d[0] = pb.q[0]; d[1] = pb.q[1]; }
        __syncthreads();

        // ---- prefetch tile k0+32 into registers (latency hidden by WMMAs) ----
        if (k0 + 32 < D_MODEL) {
            const float* an = aseg + k0 + 32;
            #pragma unroll
            for (int i = 0; i < 4; ++i) {
                float4 v = *(const float4*)(an + i * 4);
                areg[4*i+0] = v.x; areg[4*i+1] = v.y; areg[4*i+2] = v.z; areg[4*i+3] = v.w;
            }
            const float* bnp = bseg + (size_t)(k0 + 32) * D_FF;
            #pragma unroll
            for (int j = 0; j < 16; ++j) breg[j] = bnp[(size_t)j * D_FF];
        }

        // ---- fragments in contiguous k order (K = 16*half + e), then 8 WMMAs ----
        union Frag { v16bf v; uint4 q[2]; };
        Frag a[2], b[4];
        #pragma unroll
        for (int tm = 0; tm < 2; ++tm) {
            int m = wm * 32 + tm * 16 + l16;
            const uint4* p = (const uint4*)&lsA[m * LDA];
            a[tm].q[0] = p[2 * half];        // K = 16*half + e
            a[tm].q[1] = p[2 * half + 1];    // K = 16*half + 8 + e
        }
        #pragma unroll
        for (int tn = 0; tn < 4; ++tn) {
            int n = wn * 64 + tn * 16 + l16;
            const uint4* p = (const uint4*)&lsB[n * LDA + 16 * half];  // K = 16*half + e
            b[tn].q[0] = p[0];
            b[tn].q[1] = p[1];
        }
        #pragma unroll
        for (int tm = 0; tm < 2; ++tm)
            #pragma unroll
            for (int tn = 0; tn < 4; ++tn)
                acc[tm][tn] = __builtin_amdgcn_wmma_f32_16x16x32_bf16(
                    false, a[tm].v, false, b[tn].v, (short)0, acc[tm][tn], false, false);
        __syncthreads();
    }

    // ---- epilogue: bias + exact GELU -> bf16 H ----
    #pragma unroll
    for (int tm = 0; tm < 2; ++tm)
        #pragma unroll
        for (int tn = 0; tn < 4; ++tn) {
            int n = bn + wn * 64 + tn * 16 + l16;
            float bias = b1[n];
            #pragma unroll
            for (int r = 0; r < 8; ++r) {
                int m = bm + wm * 32 + tm * 16 + r + 8 * half;
                H[(size_t)m * D_FF + n] = (bf16_t)gelu_exact(acc[tm][tn][r] + bias);
            }
        }
}

// ---------------------------------------------------------------------------
// GEMM2: out[m,d] = H[m,:] @ W2[:,d] + b2[d].  A (H) is already bf16 ->
// stage it with CDNA5 async global->LDS copies (ASYNCcnt), double-buffered.
// ---------------------------------------------------------------------------
__global__ __launch_bounds__(256)
void ffn_gemm2(const bf16_t* __restrict__ H, const float* __restrict__ W2,
               const float* __restrict__ b2, float* __restrict__ out)
{
    __shared__ bf16_t lsA[2][128 * LDA];
    __shared__ bf16_t lsB[128 * LDA];

    const int t    = threadIdx.x;
    const int lane = t & 31;
    const int wave = t >> 5;
    const int wm   = wave & 3;
    const int wn   = wave >> 2;
    const int half = lane >> 4;
    const int l16  = lane & 15;
    const int bm   = blockIdx.y * 128;
    const int bn   = blockIdx.x * 128;

    const int arow = t >> 1;
    const int acb  = (t & 1) * 16;
    const int bcol = t & 127;
    const int bkh  = (t >> 7) * 16;

    const bf16_t* asrc = H  + (size_t)(bm + arow) * D_FF + acb;     // + k0
    const float*  bseg = W2 + (size_t)bkh * D_MODEL + bn + bcol;    // + k0*D_MODEL

    v8f acc[2][4];
    #pragma unroll
    for (int i = 0; i < 2; ++i)
        #pragma unroll
        for (int j = 0; j < 4; ++j) acc[i][j] = (v8f){0,0,0,0,0,0,0,0};

    float breg[16];
#ifndef HAVE_ASYNC_LDS
    uint4 areg[2];
#endif

    // prologue: tile 0
#ifdef HAVE_ASYNC_LDS
    __builtin_amdgcn_global_load_async_to_lds_b128(
        (gv4_t)asrc, (lv4_t)&lsA[0][arow * LDA + acb], 0, 0);
    __builtin_amdgcn_global_load_async_to_lds_b128(
        (gv4_t)asrc, (lv4_t)&lsA[0][arow * LDA + acb], 16, 0);
#else
    areg[0] = ((const uint4*)asrc)[0];
    areg[1] = ((const uint4*)asrc)[1];
#endif
    #pragma unroll
    for (int j = 0; j < 16; ++j) breg[j] = bseg[(size_t)j * D_MODEL];

    int buf = 0;
    for (int k0 = 0; k0 < D_FF; k0 += 32) {
        const bool more = (k0 + 32 < D_FF);

        // ---- B: convert + packed stores ----
        union Pack { bf16_t h[16]; uint4 q[2]; } pb;
        #pragma unroll
        for (int i = 0; i < 16; ++i) pb.h[i] = (bf16_t)breg[i];
        { uint4* d = (uint4*)&lsB[bcol * LDA + bkh]; d[0] = pb.q[0]; d[1] = pb.q[1]; }

#ifdef HAVE_ASYNC_LDS
        // ---- A: issue async copy of NEXT tile, then retire CURRENT tile's copies ----
        if (more) {
            __builtin_amdgcn_global_load_async_to_lds_b128(
                (gv4_t)(asrc + k0 + 32), (lv4_t)&lsA[buf ^ 1][arow * LDA + acb], 0, 0);
            __builtin_amdgcn_global_load_async_to_lds_b128(
                (gv4_t)(asrc + k0 + 32), (lv4_t)&lsA[buf ^ 1][arow * LDA + acb], 16, 0);
            __builtin_amdgcn_s_wait_asynccnt(2);   // keep only next-tile copies in flight
        } else {
            __builtin_amdgcn_s_wait_asynccnt(0);
        }
#else
        { uint4* d = (uint4*)&lsA[buf][arow * LDA + acb]; d[0] = areg[0]; d[1] = areg[1]; }
#endif
        __syncthreads();

        // ---- prefetch next B (and fallback A) into registers ----
        if (more) {
            const float* bnp = bseg + (size_t)(k0 + 32) * D_MODEL;
            #pragma unroll
            for (int j = 0; j < 16; ++j) breg[j] = bnp[(size_t)j * D_MODEL];
#ifndef HAVE_ASYNC_LDS
            const uint4* anp = (const uint4*)(asrc + k0 + 32);
            areg[0] = anp[0];
            areg[1] = anp[1];
#endif
        }

        // ---- fragments + 8 WMMAs ----
        union Frag { v16bf v; uint4 q[2]; };
        Frag a[2], b[4];
        #pragma unroll
        for (int tm = 0; tm < 2; ++tm) {
            int m = wm * 32 + tm * 16 + l16;
            const uint4* p = (const uint4*)&lsA[buf][m * LDA];
            a[tm].q[0] = p[2 * half];
            a[tm].q[1] = p[2 * half + 1];
        }
        #pragma unroll
        for (int tn = 0; tn < 4; ++tn) {
            int n = wn * 64 + tn * 16 + l16;
            const uint4* p = (const uint4*)&lsB[n * LDA + 16 * half];
            b[tn].q[0] = p[0];
            b[tn].q[1] = p[1];
        }
        #pragma unroll
        for (int tm = 0; tm < 2; ++tm)
            #pragma unroll
            for (int tn = 0; tn < 4; ++tn)
                acc[tm][tn] = __builtin_amdgcn_wmma_f32_16x16x32_bf16(
                    false, a[tm].v, false, b[tn].v, (short)0, acc[tm][tn], false, false);
        __syncthreads();
        buf ^= 1;
    }

    #pragma unroll
    for (int tm = 0; tm < 2; ++tm)
        #pragma unroll
        for (int tn = 0; tn < 4; ++tn) {
            int n = bn + wn * 64 + tn * 16 + l16;
            float bias = b2[n];
            #pragma unroll
            for (int r = 0; r < 8; ++r) {
                int m    = bm + wm * 32 + tm * 16 + r + 8 * half;
                int orow = (m >> 10) * SEQ_LEN + (m & 1023);
                out[(size_t)orow * D_MODEL + n] = acc[tm][tn][r] + bias;
            }
        }
}

// ---------------------------------------------------------------------------
// NS experts, stage 1: hn[e,b,f] = GELU(dot(x_ns[b,e,:], W1n[e,:,f]) + b1n[e,f])
// HBM-bound on W1n (268 MB): stream rows contiguously, broadcast x from LDS.
// ---------------------------------------------------------------------------
__global__ __launch_bounds__(256)
void ns_ffn1(const float* __restrict__ x, const float* __restrict__ W1n,
             const float* __restrict__ b1n, float* __restrict__ hn)
{
    __shared__ float xs[BATCH * D_MODEL];   // 32 KB
    const int e = blockIdx.x;
    const int f = blockIdx.y * 256 + threadIdx.x;

    for (int i = threadIdx.x; i < BATCH * D_MODEL; i += 256) {
        int b = i >> 10, d = i & 1023;
        xs[i] = x[((size_t)b * SEQ_LEN + SEQ_TOK + e) * D_MODEL + d];
    }
    __syncthreads();

    float acc[BATCH] = {};
    const float* w = W1n + (size_t)e * D_MODEL * D_FF + f;
    for (int d = 0; d < D_MODEL; ++d) {
        float wv = w[(size_t)d * D_FF];
        #pragma unroll
        for (int b = 0; b < BATCH; ++b) acc[b] += xs[b * D_MODEL + d] * wv;
    }
    float bias = b1n[e * D_FF + f];
    #pragma unroll
    for (int b = 0; b < BATCH; ++b)
        hn[((size_t)e * BATCH + b) * D_FF + f] = gelu_exact(acc[b] + bias);
}

// ---------------------------------------------------------------------------
// NS experts, stage 2: out[b, 1024+e, d] = dot(hn[e,b,:], W2n[e,:,d]) + b2n[e,d]
// ---------------------------------------------------------------------------
__global__ __launch_bounds__(256)
void ns_ffn2(const float* __restrict__ hn, const float* __restrict__ W2n,
             const float* __restrict__ b2n, float* __restrict__ out)
{
    __shared__ float hs[BATCH * 512];       // 16 KB chunk
    const int e = blockIdx.x;
    const int d = blockIdx.y * 256 + threadIdx.x;

    float acc[BATCH] = {};
    for (int fc = 0; fc < D_FF; fc += 512) {
        for (int i = threadIdx.x; i < BATCH * 512; i += 256) {
            int b = i >> 9, f = i & 511;
            hs[i] = hn[((size_t)e * BATCH + b) * D_FF + fc + f];
        }
        __syncthreads();
        for (int f = 0; f < 512; ++f) {
            float wv = W2n[((size_t)e * D_FF + fc + f) * D_MODEL + d];
            #pragma unroll
            for (int b = 0; b < BATCH; ++b) acc[b] += hs[b * 512 + f] * wv;
        }
        __syncthreads();
    }
    float bias = b2n[e * D_MODEL + d];
    #pragma unroll
    for (int b = 0; b < BATCH; ++b)
        out[((size_t)b * SEQ_LEN + SEQ_TOK + e) * D_MODEL + d] = acc[b] + bias;
}

extern "C" void kernel_launch(void* const* d_in, const int* in_sizes, int n_in,
                              void* d_out, int out_size, void* d_ws, size_t ws_size,
                              hipStream_t stream) {
    const float* x   = (const float*)d_in[0];
    const float* W1  = (const float*)d_in[1];
    const float* b1  = (const float*)d_in[2];
    const float* W2  = (const float*)d_in[3];
    const float* b2  = (const float*)d_in[4];
    const float* W1n = (const float*)d_in[5];
    const float* b1n = (const float*)d_in[6];
    const float* W2n = (const float*)d_in[7];
    const float* b2n = (const float*)d_in[8];
    float* out = (float*)d_out;

    // workspace: H bf16 (64 MB) | hn f32 (2 MB)
    bf16_t* H  = (bf16_t*)d_ws;
    float*  hn = (float*)((char*)d_ws + (size_t)M_SEQ * D_FF * sizeof(bf16_t));

    dim3 g1(D_FF / 128, M_SEQ / 128);      // (32, 64)
    ffn_gemm1<<<g1, 256, 0, stream>>>(x, W1, b1, H);

    dim3 g2(D_MODEL / 128, M_SEQ / 128);   // (8, 64)
    ffn_gemm2<<<g2, 256, 0, stream>>>(H, W2, b2, out);

    ns_ffn1<<<dim3(NS_TOK, D_FF / 256), 256, 0, stream>>>(x, W1n, b1n, hn);
    ns_ffn2<<<dim3(NS_TOK, D_MODEL / 256), 256, 0, stream>>>(hn, W2n, b2n, out);
}
